// MultiHeadAttention_2207613190505
// MI455X (gfx1250) — matrix operand order repaired and checked by reading, not compiled, not run on hardware
//
#include <hip/hip_runtime.h>

// MI455X (gfx1250) causal GQA attention layer.
// bf16 WMMA compute; fragment-order-swizzled, double-buffered LDS tiles;
// async global->LDS (ASYNCcnt) staging overlapped with WMMA compute.
// B=1, S=2048, H=4096, NH=32, NKV=8, HD=128.

typedef __bf16 bf16;
typedef __attribute__((ext_vector_type(2)))  __bf16        bf16x2;
typedef __attribute__((ext_vector_type(16))) __bf16        v16bf;
typedef __attribute__((ext_vector_type(8)))  float         v8f;
typedef __attribute__((ext_vector_type(8)))  unsigned int  v8u;
typedef __attribute__((ext_vector_type(4)))  float         f32x4;
typedef __attribute__((ext_vector_type(4)))  unsigned int  u32x4;

#define S_LEN 2048
#define NHQ   32
#define NKVH  8
#define HD    128
#define NREP  4
#define TK    32

// ---- ISA fragment K-index maps (CDNA5 ISA §7.12.2) ----
__device__ __forceinline__ int kkA(int v, int hf) {            // A elem pair 2v,2v+1
  return ((v & 4) ? 16 : 0) + hf * 8 + (v & 3) * 2;
}
__device__ __forceinline__ int hfA_of(int k) { return (k >> 3) & 1; }
__device__ __forceinline__ int eA_of(int k) {
  return 2 * (((k >> 4) << 2) + ((k & 7) >> 1)) + (k & 1);
}
// B 16-bit 32x16: lane%16=N; K order same as A (table order via hfA_of/eA_of)

// single v_cvt_pk_bf16_f32
__device__ __forceinline__ unsigned pack2f(float a, float b) {
  bf16x2 p;
  p[0] = (bf16)a;
  p[1] = (bf16)b;
  return __builtin_bit_cast(unsigned, p);
}

// ---------------------------------------------------------------------
// GEMM: C[M,N] = A[M,K] * B[K,N]; f32 in memory, bf16 WMMA compute.
// Block tile 128x128, K-step 32, 8 waves; double-buffered LDS, pipelined:
//   iter i: issue global loads for tile i+1 -> WMMA on tile i -> cvt+store i+1.
// ---------------------------------------------------------------------
__global__ __launch_bounds__(256) void gemm_f32_bf16wmma(
    const float* __restrict__ A, const float* __restrict__ B,
    float* __restrict__ C, int M, int N, int K)
{
  __shared__ __align__(32) bf16 sAs[2][8][32][16];  // [buf][row-tile][lane][elem]
  __shared__ __align__(32) bf16 sBs[2][8][32][16];  // [buf][n-tile][lane][elem]
  const int tid  = threadIdx.x;
  const int wave = tid >> 5;
  const int lane = tid & 31;
  const int hf   = lane >> 4;
  const int ln   = lane & 15;
  const int m0   = blockIdx.x * 128;
  const int n0   = blockIdx.y * 128;

  // fixed per-thread staging assignment
  const int ar  = tid >> 1;           // A: row 0..127
  const int ak  = (tid & 1) * 16;     // A: k half-chunk
  const int brp = tid >> 4;           // B: row pair (rows 2brp, 2brp+1)
  const int bcc = (tid & 15) * 8;     // B: col chunk of 8

  auto loadTiles = [&](int k0, f32x4* fa, f32x4* fb0, f32x4* fb1) {
    const float* ap = &A[(size_t)(m0 + ar) * K + (k0 + ak)];
#pragma unroll
    for (int q = 0; q < 4; ++q) fa[q] = *(const f32x4*)(ap + 4 * q);
    const float* bp0 = &B[(size_t)(k0 + brp * 2) * N + (n0 + bcc)];
    const float* bp1 = bp0 + N;
#pragma unroll
    for (int q = 0; q < 2; ++q) { fb0[q] = *(const f32x4*)(bp0 + 4 * q);
                                  fb1[q] = *(const f32x4*)(bp1 + 4 * q); }
  };
  auto storeTiles = [&](int buf, const f32x4* fa, const f32x4* fb0, const f32x4* fb1) {
    const float* fv = (const float*)fa;
#pragma unroll
    for (int j = 0; j < 16; j += 2) {
      int k = ak + j;                                     // even
      int slot = hfA_of(k) * 16 + (ar & 15);
      *(unsigned*)&sAs[buf][ar >> 4][slot][eA_of(k)] = pack2f(fv[j], fv[j + 1]);
    }
    const float* r0 = (const float*)fb0;
    const float* r1 = (const float*)fb1;
    int bhf = hfA_of(brp * 2), be = eA_of(brp * 2);       // even
#pragma unroll
    for (int j = 0; j < 8; ++j) {
      int n = bcc + j;
      *(unsigned*)&sBs[buf][n >> 4][bhf * 16 + (n & 15)][be] = pack2f(r0[j], r1[j]);
    }
  };

  v8f acc[8];
#pragma unroll
  for (int t = 0; t < 8; ++t) acc[t] = (v8f)0.0f;

  // prologue: stage tile 0
  {
    f32x4 fa[4], fb0[2], fb1[2];
    loadTiles(0, fa, fb0, fb1);
    storeTiles(0, fa, fb0, fb1);
  }
  __syncthreads();

  for (int k0 = 0; k0 < K; k0 += TK) {
    const int cur = (k0 >> 5) & 1;
    const bool more = (k0 + TK) < K;
    f32x4 fa[4], fb0[2], fb1[2];
    if (more) {
      loadTiles(k0 + TK, fa, fb0, fb1);   // global_load_b128s in flight over WMMAs
      if (k0 + 2 * TK < K) {
        __builtin_prefetch(&A[(size_t)(m0 + ar) * K + k0 + 2 * TK + ak], 0, 1);
        __builtin_prefetch(&B[(size_t)(k0 + 2 * TK + brp) * N + n0 + bcc], 0, 1);
      }
    }

    v16bf afr = *(const v16bf*)&sAs[cur][wave][lane][0];
#pragma unroll
    for (int t = 0; t < 8; ++t) {
      v16bf bfr = *(const v16bf*)&sBs[cur][t][lane][0];
      acc[t] = __builtin_amdgcn_wmma_f32_16x16x32_bf16(
          false, afr, false, bfr, (short)0, acc[t], false, false);
    }

    if (more) storeTiles(cur ^ 1, fa, fb0, fb1);  // s_wait_loadcnt lands here
    __syncthreads();
  }

#pragma unroll
  for (int t = 0; t < 8; ++t)
#pragma unroll
    for (int r = 0; r < 8; ++r) {
      int m = m0 + wave * 16 + r + 8 * hf;   // C layout: M = vgpr + 8*half
      int n = n0 + t * 16 + ln;              //           N = lane%16
      C[(size_t)m * N + n] = acc[t][r];
    }
}

// ---------------------------------------------------------------------
// RoPE + pack to head-major bf16. Q gets 1/sqrt(HD) folded in.
// ---------------------------------------------------------------------
__global__ __launch_bounds__(256) void rope_pack(
    const float* __restrict__ Qf, const float* __restrict__ Kf,
    const float* __restrict__ Vf, const float* __restrict__ cosT,
    const float* __restrict__ sinT,
    bf16* __restrict__ Qb, bf16* __restrict__ Kb, bf16* __restrict__ Vb)
{
  const int s = blockIdx.x;
  const float qscale = 0.08838834764831845f; // 1/sqrt(128)
  for (int i = threadIdx.x; i < NHQ * HD; i += 256) {
    int h = i >> 7, d = i & 127;
    float c = cosT[s * HD + d], sn = sinT[s * HD + d];
    float x = Qf[(size_t)s * (NHQ * HD) + i];
    float xr = (d < 64) ? -Qf[(size_t)s * (NHQ * HD) + h * HD + d + 64]
                        :  Qf[(size_t)s * (NHQ * HD) + h * HD + d - 64];
    Qb[((size_t)h * S_LEN + s) * HD + d] = (bf16)((x * c + xr * sn) * qscale);
  }
  for (int i = threadIdx.x; i < NKVH * HD; i += 256) {
    int h = i >> 7, d = i & 127;
    float c = cosT[s * HD + d], sn = sinT[s * HD + d];
    float x = Kf[(size_t)s * (NKVH * HD) + i];
    float xr = (d < 64) ? -Kf[(size_t)s * (NKVH * HD) + h * HD + d + 64]
                        :  Kf[(size_t)s * (NKVH * HD) + h * HD + d - 64];
    Kb[((size_t)h * S_LEN + s) * HD + d] = (bf16)(x * c + xr * sn);
    Vb[((size_t)h * S_LEN + s) * HD + d] = (bf16)Vf[(size_t)s * (NKVH * HD) + i];
  }
}

// ---------------------------------------------------------------------
// Flash-style causal attention; bf16 WMMA for QK^T and PV.
// grid = (S/128, NH); 8 waves, wave w owns queries [q0,q0+16).
// Double-buffered K/V tiles: K via async global->LDS DMA (issued for chunk
// i+1 before chunk i's WMMAs, s_wait_asynccnt after), V via reg-staged
// v_perm_b32-packed ds_store_b32 placed after the compute.
// ---------------------------------------------------------------------
__global__ __launch_bounds__(256) void flash_attn_wmma(
    const bf16* __restrict__ Qb, const bf16* __restrict__ Kb,
    const bf16* __restrict__ Vb, float* __restrict__ Out)
{
  __shared__ __align__(32) bf16 sKs[2][2][4][32][16]; // [buf][kt][c][lane][e] 16KB
  __shared__ __align__(32) bf16 sVs[2][8][32][16];    // [buf][t][lane][e]    16KB
  __shared__ __align__(32) bf16 sPs[8][32][16];       // [wave][lane][e]       8KB

  const int tid  = threadIdx.x;
  const int h    = blockIdx.y;
  const int hkv  = h / NREP;
  const int wave = tid >> 5;
  const int lane = tid & 31;
  const int hf   = lane >> 4;
  const int ln   = lane & 15;
  const int q0   = blockIdx.x * 128 + wave * 16;

  // Q fragments: 16 queries x 128 d, 4 chunks of K=32, dword global loads
  v16bf qf[4];
  {
    const unsigned* qrow32 =
        (const unsigned*)(Qb + ((size_t)h * S_LEN + (q0 + ln)) * HD);
#pragma unroll
    for (int c = 0; c < 4; ++c) {
      v8u tq;
#pragma unroll
      for (int v = 0; v < 8; ++v) tq[v] = qrow32[(c * 32 + kkA(v, hf)) >> 1];
      qf[c] = __builtin_bit_cast(v16bf, tq);
    }
  }

  v8f oacc[8];
#pragma unroll
  for (int t = 0; t < 8; ++t) oacc[t] = (v8f)0.0f;
  float mrow[8], lrow[8];
#pragma unroll
  for (int r = 0; r < 8; ++r) { mrow[r] = -3.0e38f; lrow[r] = 0.0f; }

  const unsigned sK_base = (unsigned)(size_t)&sKs[0][0][0][0][0];
  const unsigned long long kglob =
      (unsigned long long)(const void*)(Kb + (size_t)hkv * S_LEN * HD);
  const int vkey = (tid >> 4) * 2;        // V staging: key pair
  const int vcc  = (tid & 15) * 8;        // V staging: d chunk of 8
  // P scatter: hfA(ln) == hfA(ln+16); eA(ln+16) == eA(ln)+8
  const int pHf = hfA_of(ln);
  const int pE  = eA_of(ln);

  auto stageK = [&](int k0, int buf) {    // 8 async dword DMAs per thread
#pragma unroll
    for (int j = 0; j < 8; ++j) {
      int idx = tid + j * 256;
      int key = idx >> 6;
      int d   = (idx & 63) * 2;
      int kt = key >> 4, lnk = key & 15;
      int c = d >> 5, khf = hfA_of(d & 31), e = eA_of(d & 31);
      unsigned ldsoff = sK_base + (unsigned)buf * 8192u +
          (unsigned)(((((kt << 2) + c) << 5) + khf * 16 + lnk) * 16 + e) * 2u;
      unsigned goff = (unsigned)(((k0 + key) << 7) + d) * 2u;
      asm volatile("global_load_async_to_lds_b32 %0, %1, %2 offset:0"
                   :: "v"(ldsoff), "v"(goff), "s"(kglob) : "memory");
    }
  };
  auto loadV = [&](int k0, u32x4& w0, u32x4& w1) {
    const bf16* vr0 = Vb + ((size_t)hkv * S_LEN + k0 + vkey) * HD + vcc;
    w0 = *(const u32x4*)vr0;
    w1 = *(const u32x4*)(vr0 + HD);
  };
  auto storeV = [&](int buf, u32x4 w0, u32x4 w1) {
    int vhf = hfA_of(vkey), ve = eA_of(vkey);  // even
#pragma unroll
    for (int q = 0; q < 4; ++q) {         // v_perm_b32 interleave of the 2 rows
      unsigned lo = __builtin_amdgcn_perm(w1[q], w0[q], 0x05040100u);
      unsigned hi = __builtin_amdgcn_perm(w1[q], w0[q], 0x07060302u);
      int d0 = vcc + 2 * q, d1 = d0 + 1;
      *(unsigned*)&sVs[buf][d0 >> 4][vhf * 16 + (d0 & 15)][ve] = lo;
      *(unsigned*)&sVs[buf][d1 >> 4][vhf * 16 + (d1 & 15)][ve] = hi;
    }
  };

  // prologue: stage chunk 0 into buffer 0
  {
    u32x4 w0, w1;
    stageK(0, 0);
    loadV(0, w0, w1);
    storeV(0, w0, w1);
    asm volatile("s_wait_asynccnt 0x0" ::: "memory");
  }
  __syncthreads();

  const int kend = (blockIdx.x + 1) * 128;
  for (int k0 = 0; k0 < kend; k0 += 32) {
    const int cur = (k0 >> 5) & 1;
    const bool more = (k0 + 32) < kend;
    u32x4 w0, w1;
    if (more) {                 // async DMA + V loads for chunk i+1 in flight
      stageK(k0 + 32, cur ^ 1);
      loadV(k0 + 32, w0, w1);
    }

    if (k0 <= q0 + 15) {        // wave has unmasked work in this chunk
      v8f sc[2];
#pragma unroll
      for (int kt = 0; kt < 2; ++kt) {
        v8f a = (v8f)0.0f;
#pragma unroll
        for (int c = 0; c < 4; ++c) {
          v16bf bfr = *(const v16bf*)&sKs[cur][kt][c][lane][0];
          a = __builtin_amdgcn_wmma_f32_16x16x32_bf16(
              false, qf[c], false, bfr, (short)0, a, false, false);
        }
        sc[kt] = a;
      }
      // causal mask (row q0+r+8*hf, col k0+kt*16+ln)
#pragma unroll
      for (int kt = 0; kt < 2; ++kt)
#pragma unroll
        for (int r = 0; r < 8; ++r)
          if (k0 + kt * 16 + ln > q0 + r + 8 * hf) sc[kt][r] = -1.0e9f;

      // online softmax; rows live across 16 lanes of one half per VGPR
#pragma unroll
      for (int r = 0; r < 8; ++r) {
        float vmax = fmaxf(sc[0][r], sc[1][r]);
        vmax = fmaxf(vmax, __shfl_xor(vmax, 1));
        vmax = fmaxf(vmax, __shfl_xor(vmax, 2));
        vmax = fmaxf(vmax, __shfl_xor(vmax, 4));
        vmax = fmaxf(vmax, __shfl_xor(vmax, 8));
        float newm = fmaxf(mrow[r], vmax);
        float resc = __expf(mrow[r] - newm);
        mrow[r] = newm;
        float p0 = __expf(sc[0][r] - newm);
        float p1 = __expf(sc[1][r] - newm);
        bf16* prow = &sPs[wave][pHf * 16 + r + 8 * hf][0];
        prow[pE]     = (bf16)p0;
        prow[pE + 8] = (bf16)p1;
        float ps = p0 + p1;
        ps += __shfl_xor(ps, 1);
        ps += __shfl_xor(ps, 2);
        ps += __shfl_xor(ps, 4);
        ps += __shfl_xor(ps, 8);
        lrow[r] = lrow[r] * resc + ps;
#pragma unroll
        for (int t = 0; t < 8; ++t) oacc[t][r] *= resc;
      }
      // wave-local LDS RAW: DS in-order per wave; stop compiler reordering
      asm volatile("s_wait_dscnt 0x0" ::: "memory");

      v16bf pfr = *(const v16bf*)&sPs[wave][lane][0];
#pragma unroll
      for (int t = 0; t < 8; ++t) {
        v16bf bfr = *(const v16bf*)&sVs[cur][t][lane][0];
        oacc[t] = __builtin_amdgcn_wmma_f32_16x16x32_bf16(
            false, pfr, false, bfr, (short)0, oacc[t], false, false);
      }
    }

    if (more) storeV(cur ^ 1, w0, w1);
    asm volatile("s_wait_asynccnt 0x0" ::: "memory");
    __syncthreads();
  }

#pragma unroll
  for (int t = 0; t < 8; ++t)
#pragma unroll
    for (int r = 0; r < 8; ++r) {
      int q = q0 + r + 8 * hf;
      Out[(size_t)q * (NHQ * HD) + h * HD + t * 16 + ln] = oacc[t][r] / lrow[r];
    }
}

// ---------------------------------------------------------------------
extern "C" void kernel_launch(void* const* d_in, const int* in_sizes, int n_in,
                              void* d_out, int out_size, void* d_ws, size_t ws_size,
                              hipStream_t stream) {
  (void)in_sizes; (void)n_in; (void)out_size; (void)ws_size;
  const float* X    = (const float*)d_in[0];
  // d_in[1] position_ids, d_in[2] attention_mask: causal mask applied analytically
  const float* cosT = (const float*)d_in[3];
  const float* sinT = (const float*)d_in[4];
  const float* Wq   = (const float*)d_in[5];
  const float* Wk   = (const float*)d_in[6];
  const float* Wv   = (const float*)d_in[7];
  const float* Wo   = (const float*)d_in[8];
  float* out = (float*)d_out;

  char* ws = (char*)d_ws;
  float* Qf   = (float*)(ws);                 // 33554432 B
  float* Kf   = (float*)(ws + 33554432ull);   //  8388608 B
  float* Vf   = (float*)(ws + 41943040ull);   //  8388608 B
  bf16*  Qb   = (bf16*) (ws + 50331648ull);   // 16777216 B
  bf16*  Kb   = (bf16*) (ws + 67108864ull);   //  4194304 B
  bf16*  Vb   = (bf16*) (ws + 71303168ull);   //  4194304 B
  float* Attn = (float*)(ws + 75497472ull);   // 33554432 B

  gemm_f32_bf16wmma<<<dim3(16, 32), 256, 0, stream>>>(X, Wq, Qf, 2048, 4096, 4096);
  gemm_f32_bf16wmma<<<dim3(16, 8),  256, 0, stream>>>(X, Wk, Kf, 2048, 1024, 4096);
  gemm_f32_bf16wmma<<<dim3(16, 8),  256, 0, stream>>>(X, Wv, Vf, 2048, 1024, 4096);
  rope_pack<<<dim3(S_LEN), 256, 0, stream>>>(Qf, Kf, Vf, cosT, sinT, Qb, Kb, Vb);
  flash_attn_wmma<<<dim3(16, 32), 256, 0, stream>>>(Qb, Kb, Vb, Attn);
  gemm_f32_bf16wmma<<<dim3(16, 32), 256, 0, stream>>>(Attn, Wo, out, 2048, 4096, 4096);
}
